// MyRNN_5128190952074
// MI455X (gfx1250) — matrix operand order repaired and checked by reading, not compiled, not run on hardware
//
#include <hip/hip_runtime.h>
#include <hip/hip_bf16.h>

typedef __attribute__((ext_vector_type(16))) _Float16 v16h;
typedef __attribute__((ext_vector_type(8)))  _Float16 v8h;
typedef __attribute__((ext_vector_type(4)))  _Float16 v4h;
typedef __attribute__((ext_vector_type(8)))  float    v8f;
typedef __attribute__((ext_vector_type(4)))  float    v4f;

#define RNN_B 4096
#define RNN_T 80
#define RNN_E 100
#define RNN_H 64
#define NW    2             // waves per block
#define BLOCK (NW * 32)     // wave32

// Hardware v_tanh_f32 on gfx1250; branch-free fallback otherwise.
__device__ __forceinline__ float fast_tanh(float x) {
#if __has_builtin(__builtin_amdgcn_tanhf)
    return __builtin_amdgcn_tanhf(x);
#else
    float e = __expf(2.0f * x);
    return 1.0f - 2.0f * __builtin_amdgcn_rcpf(e + 1.0f);
#endif
}

__device__ __forceinline__ float fast_sigmoid(float x) {
    return __builtin_amdgcn_rcpf(1.0f + __expf(-x));
}

// ---------------------------------------------------------------------------
// A-matrix fragment load (16x32 f16, ISA 7.12.2): lane l holds row M = l&15;
// half = l>>4; e<8 -> K = half*8+e ; e>=8 -> K = 16+half*8+(e-8).
// ---------------------------------------------------------------------------
__device__ __forceinline__ v16h ldsA(const _Float16* base, int lane, int kc, int stride) {
    const _Float16* p = base + (lane & 15) * stride + kc * 32 + (lane >> 4) * 8;
    union { v16h v; v8h h[2]; } u;
    u.h[0] = *(const v8h*)(p);
    u.h[1] = *(const v8h*)(p + 16);
    return u.v;
}

// B fragments pre-packed linearly: frag f, lane l -> 16 contiguous f16.
__device__ __forceinline__ v16h ldsB(const _Float16* frags, int lane, int fragIdx) {
    return *(const v16h*)(frags + fragIdx * 512 + lane * 16);
}

// ---------------------------------------------------------------------------
// Pack a [Krows x 64] f32 weight into WMMA B-fragment order (f16) in LDS.
// B-frag (32x16): element e of lane l is (K = e<8 ? (l>>4)*8+e : 16+(l>>4)*8+(e-8), N = l&15).
// Row K == Krows optionally holds a bias row (pairs with x[K]=1.0 padding).
// ---------------------------------------------------------------------------
__device__ void packB(_Float16* dst, const float* __restrict__ src,
                      int Krows, int kchunks, int tid,
                      const float* __restrict__ biasrow) {
    const int total = kchunks * 4 * 512;
    for (int i = tid; i < total; i += BLOCK) {
        int f   = i >> 9;
        int rem = i & 511;
        int l   = rem >> 4;
        int e   = rem & 15;
        int kc  = f >> 2;
        int nt  = f & 3;
        int K   = kc * 32 + (e < 8 ? ((l >> 4) << 3) + e : 16 + ((l >> 4) << 3) + (e - 8));
        int N   = nt * 16 + (l & 15);
        float v = 0.0f;
        if (K < Krows)                    v = src[K * RNN_H + N];
        else if (biasrow && K == Krows)   v = biasrow[N];
        dst[i] = (_Float16)v;
    }
}

__global__ __launch_bounds__(BLOCK)
void rnn2_fused_kernel(const int*   __restrict__ inputs,
                       const float* __restrict__ emb,
                       const float* __restrict__ Wx0,
                       const float* __restrict__ Wh0,
                       const float* __restrict__ b0,
                       const float* __restrict__ Wx1,
                       const float* __restrict__ Wh1,
                       const float* __restrict__ b1,
                       const float* __restrict__ Wf,
                       const float* __restrict__ bf,
                       float*       __restrict__ out) {
    // Weight fragments (block-shared, written once): 40 frags * 1KB = 40KB
    __shared__ __align__(32) _Float16 sWx0[16 * 512];
    __shared__ __align__(32) _Float16 sWh0[ 8 * 512];
    __shared__ __align__(32) _Float16 sWx1[ 8 * 512];
    __shared__ __align__(32) _Float16 sWh1[ 8 * 512];
    // Per-wave staging
    __shared__ __align__(32) _Float16 sX [NW][16 * 128];  // x_t, K padded to 128
    __shared__ __align__(32) _Float16 sH0[NW][16 * 64];
    __shared__ __align__(32) _Float16 sH1[NW][16 * 64];
    __shared__ int sTok[NW * 16 * RNN_T];                 // block's token slab

    const int tid = threadIdx.x;
    packB(sWx0, Wx0, RNN_E, 4, tid, b0);      // K==100 row carries b0
    packB(sWh0, Wh0, RNN_H, 2, tid, nullptr);
    packB(sWx1, Wx1, RNN_H, 2, tid, nullptr);
    packB(sWh1, Wh1, RNN_H, 2, tid, nullptr);
    // This block's 32 token rows are contiguous in inputs: one coalesced copy.
    {
        const int base = blockIdx.x * (NW * 16) * RNN_T;
        for (int i = tid; i < NW * 16 * RNN_T; i += BLOCK)
            sTok[i] = inputs[base + i];
    }
    __syncthreads();

    const int lane    = tid & 31;
    const int wave    = tid >> 5;
    const int rowbase = blockIdx.x * (NW * 16) + wave * 16;
    const int nlo     = lane & 15;
    const int mbase   = (lane >> 4) * 8;      // C/D layout: VGPR v -> M = mbase+v

    _Float16* xw  = sX[wave];
    _Float16* h0w = sH0[wave];
    _Float16* h1w = sH1[wave];
    const int* tokw = sTok + wave * 16 * RNN_T;

    // One-time init: h0 = h1 = 0; x K-tail (100..127) = {1, 0, ...} per row,
    // pairing with the b0 row packed at K==100 of sWx0.
    for (int i = lane; i < 16 * 128; i += 32) xw[i] = (_Float16)0.0f;
    for (int i = lane; i < 16 * 64; i += 32) {
        h0w[i] = (_Float16)0.0f;
        h1w[i] = (_Float16)0.0f;
    }
    if (lane < 16) xw[lane * 128 + RNN_E] = (_Float16)1.0f;

    // b1 splat values (layer-1 bias; depends only on N).
    float b1v[4];
#pragma unroll
    for (int nt = 0; nt < 4; ++nt) b1v[nt] = b1[nt * 16 + nlo];

    // Software pipeline: gather x_{t+1} into registers while computing step t.
    v4f xreg[16];
    {
        int tok = tokw[nlo * RNN_T + 0];
        if (lane < 25) {
#pragma unroll
            for (int r = 0; r < 16; ++r) {
                int tr = __builtin_amdgcn_readlane(tok, r);
                xreg[r] = *(const v4f*)(emb + (size_t)tr * RNN_E + lane * 4);
            }
        }
    }

    for (int t = 0; t < RNN_T; ++t) {
        // Commit x_t (cvt f32->f16, one ds_store_b64 per row per lane<25).
        if (lane < 25) {
#pragma unroll
            for (int r = 0; r < 16; ++r) {
                v4h h;
#pragma unroll
                for (int j = 0; j < 4; ++j) h[j] = (_Float16)xreg[r][j];
                *(v4h*)(xw + r * 128 + lane * 4) = h;
            }
        }

        // ---------------- layer 0: z0 = [x,1]@[Wx0;b0] + h0@Wh0 ----------------
        v8f acc[4];
#pragma unroll
        for (int nt = 0; nt < 4; ++nt) {
            v8f z = {0.f, 0.f, 0.f, 0.f, 0.f, 0.f, 0.f, 0.f};
            acc[nt] = z;     // inline SRC2=0 on first WMMA
        }
#pragma unroll
        for (int kc = 0; kc < 4; ++kc) {
            v16h a = ldsA(xw, lane, kc, 128);
#pragma unroll
            for (int nt = 0; nt < 4; ++nt)
                acc[nt] = __builtin_amdgcn_wmma_f32_16x16x32_f16(
                    false, a, false, ldsB(sWx0, lane, kc * 4 + nt),
                    (short)0, acc[nt], false, false);
        }
#pragma unroll
        for (int kc = 0; kc < 2; ++kc) {
            v16h a = ldsA(h0w, lane, kc, 64);
#pragma unroll
            for (int nt = 0; nt < 4; ++nt)
                acc[nt] = __builtin_amdgcn_wmma_f32_16x16x32_f16(
                    false, a, false, ldsB(sWh0, lane, kc * 4 + nt),
                    (short)0, acc[nt], false, false);
        }

        // Issue next step's embedding gather here: independent of the WMMA
        // results, so its VALU/address work can fill the WMMA->VALU hazard
        // window, and the global-load latency hides behind layer 1 + commit.
        // Clamped index (no conditional) keeps a single loop body (no peel).
        {
            int tn = (t + 1 < RNN_T) ? (t + 1) : (RNN_T - 1);
            int tok = tokw[nlo * RNN_T + tn];
            if (lane < 25) {
#pragma unroll
                for (int r = 0; r < 16; ++r) {
                    int tr = __builtin_amdgcn_readlane(tok, r);
                    xreg[r] = *(const v4f*)(emb + (size_t)tr * RNN_E + lane * 4);
                }
            }
        }

        // Batched tanh (keeps >=8 independent TRANS ops in flight), then store.
        float th[4][8];
#pragma unroll
        for (int nt = 0; nt < 4; ++nt)
#pragma unroll
            for (int v = 0; v < 8; ++v) th[nt][v] = fast_tanh(acc[nt][v]);
#pragma unroll
        for (int nt = 0; nt < 4; ++nt)
#pragma unroll
            for (int v = 0; v < 8; ++v)
                h0w[(mbase + v) * 64 + nt * 16 + nlo] = (_Float16)th[nt][v];

        // ---------------- layer 1: z1 = h0@Wx1 + h1@Wh1 + b1 ----------------
#pragma unroll
        for (int nt = 0; nt < 4; ++nt) {
            v8f a;
#pragma unroll
            for (int v = 0; v < 8; ++v) a[v] = b1v[nt];
            acc[nt] = a;
        }
#pragma unroll
        for (int kc = 0; kc < 2; ++kc) {
            v16h a = ldsA(h0w, lane, kc, 64);
#pragma unroll
            for (int nt = 0; nt < 4; ++nt)
                acc[nt] = __builtin_amdgcn_wmma_f32_16x16x32_f16(
                    false, a, false, ldsB(sWx1, lane, kc * 4 + nt),
                    (short)0, acc[nt], false, false);
        }
#pragma unroll
        for (int kc = 0; kc < 2; ++kc) {
            v16h a = ldsA(h1w, lane, kc, 64);
#pragma unroll
            for (int nt = 0; nt < 4; ++nt)
                acc[nt] = __builtin_amdgcn_wmma_f32_16x16x32_f16(
                    false, a, false, ldsB(sWh1, lane, kc * 4 + nt),
                    (short)0, acc[nt], false, false);
        }
#pragma unroll
        for (int nt = 0; nt < 4; ++nt)
#pragma unroll
            for (int v = 0; v < 8; ++v) th[nt][v] = fast_tanh(acc[nt][v]);
#pragma unroll
        for (int nt = 0; nt < 4; ++nt)
#pragma unroll
            for (int v = 0; v < 8; ++v)
                h1w[(mbase + v) * 64 + nt * 16 + nlo] = (_Float16)th[nt][v];
    }

    // out = sigmoid(h1_last @ Wf + bf), one row per lane 0..15
    if (lane < 16) {
        int row = rowbase + lane;
        float s = bf[0];
#pragma unroll 8
        for (int n = 0; n < RNN_H; ++n)
            s += (float)h1w[lane * 64 + n] * Wf[n];
        out[row] = fast_sigmoid(s);
    }
}

extern "C" void kernel_launch(void* const* d_in, const int* in_sizes, int n_in,
                              void* d_out, int out_size, void* d_ws, size_t ws_size,
                              hipStream_t stream) {
    const int*   inputs = (const int*)  d_in[0];
    const float* emb    = (const float*)d_in[1];
    const float* Wx0    = (const float*)d_in[2];
    const float* Wh0    = (const float*)d_in[3];
    const float* b0     = (const float*)d_in[4];
    const float* Wx1    = (const float*)d_in[5];
    const float* Wh1    = (const float*)d_in[6];
    const float* b1     = (const float*)d_in[7];
    const float* Wf     = (const float*)d_in[8];
    const float* bf     = (const float*)d_in[9];
    float* out = (float*)d_out;

    dim3 grid(RNN_B / (NW * 16));   // 128 blocks, 32 batch rows each
    dim3 block(BLOCK);              // 2 waves of 32
    hipLaunchKernelGGL(rnn2_fused_kernel, grid, block, 0, stream,
                       inputs, emb, Wx0, Wh0, b0, Wx1, Wh1, b1, Wf, bf, out);
}
